// PointerNet_74131135529979
// MI455X (gfx1250) — compile-verified, matrix operand order checked by reading, not run on hardware
//
#include <hip/hip_runtime.h>
#include <hip/hip_bf16.h>
#include <cstdint>
#include <cstddef>

// ---------------- problem constants ----------------
constexpr int kB = 512;   // batch
constexpr int kS = 50;    // seq len
constexpr int kE = 128;   // embed dim
constexpr int kH = 128;   // hidden dim
constexpr int kBM = 32;   // batch rows per workgroup
constexpr float kTanhC = 10.0f;

typedef __attribute__((ext_vector_type(16))) _Float16 v16h;
typedef __attribute__((ext_vector_type(8)))  _Float16 v8h;
typedef __attribute__((ext_vector_type(8)))  float    v8f;

// ---------------- fast math (native gfx1250 transcendentals) ----------------
#if __has_builtin(__builtin_amdgcn_tanhf)
__device__ __forceinline__ float fast_tanh(float x) { return __builtin_amdgcn_tanhf(x); }
#elif __has_builtin(__builtin_amdgcn_tanh_f32)
__device__ __forceinline__ float fast_tanh(float x) { return __builtin_amdgcn_tanh_f32(x); }
#else
__device__ __forceinline__ float fast_tanh(float x) { return tanhf(x); }
#endif

#if __has_builtin(__builtin_amdgcn_rcpf)
__device__ __forceinline__ float fast_rcp(float x) { return __builtin_amdgcn_rcpf(x); }
#else
__device__ __forceinline__ float fast_rcp(float x) { return 1.0f / x; }
#endif

__device__ __forceinline__ float sigm(float x) {
  return fast_rcp(1.0f + __expf(-x));
}

// Defeat LICM without destroying pointer provenance: a loop-variant zero
// offset. Adding it to a weight pointer keeps the address rooted at the
// kernarg (InferAddressSpaces still emits global_load_b128, not flat_load),
// but the volatile asm makes the address loop-variant, so the 256+ VGPRs of
// loop-invariant weight fragments are never hoisted out of the recurrence
// loop and spilled to scratch.
__device__ __forceinline__ int opaque_zero() {
  int x = 0;
  asm volatile("" : "+s"(x));
  return x;
}

// ---------------- WMMA helpers ----------------
__device__ __forceinline__ v8f wmma16(v16h a, v16h b, v8f c) {
  // v_wmma_f32_16x16x32_f16 : D = A(16x32 f16) * B(32x16 f16) + C(16x16 f32)
  return __builtin_amdgcn_wmma_f32_16x16x32_f16(false, a, false, b, (short)0, c,
                                                false, false);
}

// Load a 16x32 f16 fragment (A layout; B uses identical layout with row==N)
// from a row-major [rows, ld] f16 matrix. Lane l holds row (l&15); k-halves
// per ISA §7.12.2: VGPR0-3 = K in [kh, kh+8), VGPR4-7 = K in [kh+16, kh+24),
// kh = (l>>4)*8.
__device__ __forceinline__ v16h load_frag(const _Float16* __restrict__ base,
                                          int ld, int row_base, int k_base,
                                          int lane) {
  const int r  = row_base + (lane & 15);
  const int kh = (lane >> 4) << 3;
  const _Float16* p = base + (size_t)r * ld + k_base + kh;
  v8h lo = *(const v8h*)(p);
  v8h hi = *(const v8h*)(p + 16);
  v16h out;
#pragma unroll
  for (int i = 0; i < 8; ++i) { out[i] = lo[i]; out[i + 8] = hi[i]; }
  return out;
}

__device__ __forceinline__ float rng01(uint32_t a, uint32_t b, uint32_t c) {
  uint32_t x = a * 0x9E3779B1u ^ b * 0x85EBCA77u ^ c * 0xC2B2AE3Du;
  x ^= x >> 16; x *= 0x7FEB352Du; x ^= x >> 15; x *= 0x846CA68Bu; x ^= x >> 16;
  return (float)(x >> 8) * (1.0f / 16777216.0f);
}

// gates[2][4] += X(32x128 f16, ld=ldx) * Wih^T + H(32x128 f16) * Whh^T
// Tile map: id = wave*2+tt; m = id>>3 (two 16-row blocks), j = id&7 (eight
// 16-col unit blocks). Gate g's output column block is g*128 + j*16, so all
// four gates for one (row,unit) live in identical lane/VGPR slots.
__device__ __forceinline__ void lstm_gates(const _Float16* __restrict__ X, int ldx,
                                           const _Float16* __restrict__ Hs,
                                           const _Float16* __restrict__ Wih,
                                           const _Float16* __restrict__ Whh,
                                           int wave, int lane, v8f acc[2][4]) {
#pragma unroll
  for (int tt = 0; tt < 2; ++tt) {
    const int id = wave * 2 + tt;
    const int m = id >> 3, j = id & 7;
#pragma unroll
    for (int kb = 0; kb < 4; ++kb) {
      v16h a = load_frag(X, ldx, m * 16, kb * 32, lane);
#pragma unroll
      for (int g = 0; g < 4; ++g) {
        v16h b = load_frag(Wih, kE, g * kH + j * 16, kb * 32, lane);
        acc[tt][g] = wmma16(a, b, acc[tt][g]);
      }
    }
#pragma unroll
    for (int kb = 0; kb < 4; ++kb) {
      v16h a = load_frag(Hs, kH, m * 16, kb * 32, lane);
#pragma unroll
      for (int g = 0; g < 4; ++g) {
        v16h b = load_frag(Whh, kH, g * kH + j * 16, kb * 32, lane);
        acc[tt][g] = wmma16(a, b, acc[tt][g]);
      }
    }
  }
}

// ---------------- conversion / embedding kernels ----------------
__global__ void f32_to_f16_kernel(const float* __restrict__ src,
                                  _Float16* __restrict__ dst, int n) {
  int i = blockIdx.x * blockDim.x + threadIdx.x;
  if (i < n) dst[i] = (_Float16)src[i];
}

__global__ void embed_kernel(const float* __restrict__ bin /*[B,2,S]*/,
                             const float* __restrict__ emb /*[2,E]*/,
                             _Float16* __restrict__ out /*[B,S,E]*/) {
  int i = blockIdx.x * blockDim.x + threadIdx.x;
  if (i >= kB * kS * kE) return;
  int e = i & (kE - 1);
  int s = (i >> 7) % kS;
  int b = i / (kS * kE);
  float x0 = bin[b * 2 * kS + s];
  float x1 = bin[b * 2 * kS + kS + s];
  out[i] = (_Float16)(x0 * emb[e] + x1 * emb[kE + e]);
}

__global__ void zero_flags_kernel(int* __restrict__ f, int n) {
  int i = blockIdx.x * blockDim.x + threadIdx.x;
  if (i < n) f[i] = 0;
}

// ---------------- encoder LSTM (persistent, WMMA) ----------------
__global__ __launch_bounds__(256) void enc_lstm_kernel(
    const _Float16* __restrict__ emb16 /*[B,S,E]*/,
    const _Float16* __restrict__ Wih16, const _Float16* __restrict__ Whh16,
    const float* __restrict__ bih, const float* __restrict__ bhh,
    _Float16* __restrict__ enc16 /*[B*S,H]*/,
    float* __restrict__ hN, float* __restrict__ cN /*[B,H]*/) {
  __shared__ __align__(16) _Float16 hbuf[kBM][kH];
  __shared__ float bsum[4 * kH];

  const int tid = threadIdx.x, wave = tid >> 5, lane = tid & 31;
  const int b0 = blockIdx.x * kBM;

  for (int i = tid; i < 4 * kH; i += 256) bsum[i] = bih[i] + bhh[i];
  for (int i = tid; i < kBM * kH; i += 256) (&hbuf[0][0])[i] = (_Float16)0.0f;

  const v8f vzero = {0.f, 0.f, 0.f, 0.f, 0.f, 0.f, 0.f, 0.f};
  v8f cc[2] = {vzero, vzero};

  __syncthreads();
  for (int t = 0; t < kS; ++t) {
    v8f acc[2][4];
#pragma unroll
    for (int tt = 0; tt < 2; ++tt)
#pragma unroll
      for (int g = 0; g < 4; ++g) acc[tt][g] = vzero;

    const int zo = opaque_zero();  // per-iteration: keep weight loads in-loop
    const _Float16* X = emb16 + (size_t)b0 * kS * kE + (size_t)t * kE;
    lstm_gates(X, kS * kE, &hbuf[0][0], Wih16 + zo, Whh16 + zo,
               wave, lane, acc);
    __syncthreads();  // everyone done reading old h

#pragma unroll
    for (int tt = 0; tt < 2; ++tt) {
      const int id = wave * 2 + tt, m = id >> 3, j = id & 7;
      const int jc = j * 16 + (lane & 15);
      const float bi = bsum[jc], bf = bsum[kH + jc], bg = bsum[2 * kH + jc],
                  bo = bsum[3 * kH + jc];
#pragma unroll
      for (int v = 0; v < 8; ++v) {
        const int row = m * 16 + v + ((lane >> 4) << 3);
        float iv = sigm(acc[tt][0][v] + bi);
        float fv = sigm(acc[tt][1][v] + bf);
        float gv = fast_tanh(acc[tt][2][v] + bg);
        float ov = sigm(acc[tt][3][v] + bo);
        float cv = fv * cc[tt][v] + iv * gv;
        cc[tt][v] = cv;
        float hv = ov * fast_tanh(cv);
        hbuf[row][jc] = (_Float16)hv;
        enc16[((size_t)(b0 + row) * kS + t) * kH + jc] = (_Float16)hv;
      }
    }
    __syncthreads();
  }
  // final state out
#pragma unroll
  for (int tt = 0; tt < 2; ++tt) {
    const int id = wave * 2 + tt, m = id >> 3, j = id & 7;
    const int jc = j * 16 + (lane & 15);
#pragma unroll
    for (int v = 0; v < 8; ++v) {
      const int row = m * 16 + v + ((lane >> 4) << 3);
      hN[(size_t)(b0 + row) * kH + jc] = (float)hbuf[row][jc];
      cN[(size_t)(b0 + row) * kH + jc] = cc[tt][v];
    }
  }
}

// ---------------- ref projections (WMMA GEMM, M=B*S) ----------------
__global__ __launch_bounds__(256) void refproj_kernel(
    const _Float16* __restrict__ enc16 /*[B*S,H]*/,
    const _Float16* __restrict__ Wg16, const float* __restrict__ bg,
    const _Float16* __restrict__ Wp16, const float* __restrict__ bp,
    float* __restrict__ refg, float* __restrict__ refp) {
  const int tid = threadIdx.x, wave = tid >> 5, lane = tid & 31;
  const int row0 = blockIdx.x * kBM;
  const v8f vzero = {0.f, 0.f, 0.f, 0.f, 0.f, 0.f, 0.f, 0.f};
#pragma unroll
  for (int tt = 0; tt < 2; ++tt) {
    const int id = wave * 2 + tt, m = id >> 3, j = id & 7;
    v8f ag = vzero, ap = vzero;
#pragma unroll
    for (int kb = 0; kb < 4; ++kb) {
      v16h a = load_frag(enc16, kH, row0 + m * 16, kb * 32, lane);
      v16h bG = load_frag(Wg16, kH, j * 16, kb * 32, lane);
      v16h bP = load_frag(Wp16, kH, j * 16, kb * 32, lane);
      ag = wmma16(a, bG, ag);
      ap = wmma16(a, bP, ap);
    }
    const int jc = j * 16 + (lane & 15);
#pragma unroll
    for (int v = 0; v < 8; ++v) {
      const int row = row0 + m * 16 + v + ((lane >> 4) << 3);
      refg[(size_t)row * kH + jc] = ag[v] + bg[jc];
      refp[(size_t)row * kH + jc] = ap[v] + bp[jc];
    }
  }
}

// ---------------- decoder (persistent: LSTM + attention + sampling) --------
__global__ __launch_bounds__(256) void dec_kernel(
    const _Float16* __restrict__ emb16 /*[B,S,E]*/,
    const _Float16* __restrict__ Wih16, const _Float16* __restrict__ Whh16,
    const float* __restrict__ bih, const float* __restrict__ bhh,
    const _Float16* __restrict__ gWq16, const float* __restrict__ gbq,
    const float* __restrict__ gV,
    const _Float16* __restrict__ pWq16, const float* __restrict__ pbq,
    const float* __restrict__ pV,
    const float* __restrict__ refg, const float* __restrict__ refp,
    const float* __restrict__ hN, const float* __restrict__ cN,
    const _Float16* __restrict__ start16,
    int* __restrict__ gflags, float* __restrict__ out /*probs then idx*/) {
  __shared__ __align__(16) _Float16 hbuf[kBM][kH];
  __shared__ __align__(16) _Float16 xbuf[kBM][kH];
  __shared__ __align__(16) _Float16 q2h[kBM][kH];
  __shared__ float qbuf[kBM][kH];
  __shared__ float lbuf[kBM][64];
  __shared__ float bsum[4 * kH];
  __shared__ float sVg[kH], sVp[kH];
  __shared__ unsigned char maskb[kBM][64];
  __shared__ short hist[kS + 2][kBM];
  __shared__ int curidx[kBM];
  __shared__ int localflag, gconf;

  const int tid = threadIdx.x, wave = tid >> 5, lane = tid & 31;
  const int b0 = blockIdx.x * kBM;
  float* probs_out = out;                       // [S,B,S]
  float* idx_out = out + (size_t)kS * kB * kS;  // [S,B] stored as float

  for (int i = tid; i < 4 * kH; i += 256) bsum[i] = bih[i] + bhh[i];
  for (int i = tid; i < kH; i += 256) { sVg[i] = gV[i]; sVp[i] = pV[i]; }
  for (int i = tid; i < kBM * kH; i += 256) {
    int r = i >> 7, e = i & (kH - 1);
    hbuf[r][e] = (_Float16)hN[(size_t)(b0 + r) * kH + e];
    xbuf[r][e] = start16[e];
  }
  for (int i = tid; i < kBM * 64; i += 256) (&maskb[0][0])[i] = 0;
  if (tid < kBM) curidx[tid] = 0;

  const v8f vzero = {0.f, 0.f, 0.f, 0.f, 0.f, 0.f, 0.f, 0.f};
  v8f cc[2];
#pragma unroll
  for (int tt = 0; tt < 2; ++tt) {
    const int id = wave * 2 + tt, m = id >> 3, j = id & 7;
    const int jc = j * 16 + (lane & 15);
#pragma unroll
    for (int v = 0; v < 8; ++v) {
      const int row = m * 16 + v + ((lane >> 4) << 3);
      cc[tt][v] = cN[(size_t)(b0 + row) * kH + jc];
    }
  }
  __syncthreads();

  int myidx = 0;
  for (int t = 0; t < kS; ++t) {
    // mask |= prev idx (skip at t==0), matching apply_mask_to_logits
    if (t > 0 && tid < kBM) maskb[tid][curidx[tid] & 63] = 1;
    __syncthreads();

    // ---- decoder LSTM cell ----
    v8f acc[2][4];
#pragma unroll
    for (int tt = 0; tt < 2; ++tt)
#pragma unroll
      for (int g = 0; g < 4; ++g) acc[tt][g] = vzero;
    {
      const int zo = opaque_zero();
      lstm_gates(&xbuf[0][0], kH, &hbuf[0][0], Wih16 + zo, Whh16 + zo,
                 wave, lane, acc);
    }
    __syncthreads();
#pragma unroll
    for (int tt = 0; tt < 2; ++tt) {
      const int id = wave * 2 + tt, m = id >> 3, j = id & 7;
      const int jc = j * 16 + (lane & 15);
      const float bi = bsum[jc], bf = bsum[kH + jc], bg = bsum[2 * kH + jc],
                  bo = bsum[3 * kH + jc];
#pragma unroll
      for (int v = 0; v < 8; ++v) {
        const int row = m * 16 + v + ((lane >> 4) << 3);
        float iv = sigm(acc[tt][0][v] + bi);
        float fv = sigm(acc[tt][1][v] + bf);
        float gv = fast_tanh(acc[tt][2][v] + bg);
        float ov = sigm(acc[tt][3][v] + bo);
        float cv = fv * cc[tt][v] + iv * gv;
        cc[tt][v] = cv;
        hbuf[row][jc] = (_Float16)(ov * fast_tanh(cv));
      }
    }
    __syncthreads();

    // ---- glimpse query projection: q = h @ gWq^T + gbq ----
    {
      const _Float16* Wq = gWq16 + opaque_zero();
#pragma unroll
      for (int tt = 0; tt < 2; ++tt) {
        const int id = wave * 2 + tt, m = id >> 3, j = id & 7;
        v8f aq = vzero;
#pragma unroll
        for (int kb = 0; kb < 4; ++kb) {
          v16h a = load_frag(&hbuf[0][0], kH, m * 16, kb * 32, lane);
          v16h b = load_frag(Wq, kH, j * 16, kb * 32, lane);
          aq = wmma16(a, b, aq);
        }
        const int jc = j * 16 + (lane & 15);
#pragma unroll
        for (int v = 0; v < 8; ++v)
          qbuf[m * 16 + v + ((lane >> 4) << 3)][jc] = aq[v] + gbq[jc];
      }
    }
    __syncthreads();

    // ---- glimpse logits: lg[r][s] = Vg . tanh(q[r] + refg[b,s]) ----
    for (int p = tid; p < kBM * kS; p += 256) {
      const int row = p & (kBM - 1), s = p >> 5;
      const float* rg = refg + ((size_t)(b0 + row) * kS + s) * kH;
      __builtin_prefetch(refp + ((size_t)(b0 + row) * kS + s) * kH, 0, 1);
      float acc0 = 0.f;
#pragma unroll 8
      for (int hh = 0; hh < kH; ++hh)
        acc0 += sVg[hh] * fast_tanh(qbuf[row][hh] + rg[hh]);
      lbuf[row][s] = maskb[row][s] ? -3.0e38f : acc0;
    }
    __syncthreads();
    // glimpse softmax per row
    if (tid < kBM) {
      float mx = -3.0e38f;
      for (int s = 0; s < kS; ++s) mx = fmaxf(mx, lbuf[tid][s]);
      float sum = 0.f;
      for (int s = 0; s < kS; ++s) {
        float e = __expf(lbuf[tid][s] - mx); lbuf[tid][s] = e; sum += e;
      }
      float inv = fast_rcp(sum);
      for (int s = 0; s < kS; ++s) lbuf[tid][s] *= inv;
    }
    __syncthreads();
    // glimpse read-out: q2[r][hh] = sum_s w[r][s] * refg[b,s,hh]
    for (int p = tid; p < kBM * kH; p += 256) {
      const int row = p >> 7, hh = p & (kH - 1);
      const float* rg = refg + (size_t)(b0 + row) * kS * kH + hh;
      float acc0 = 0.f;
      for (int s = 0; s < kS; ++s) acc0 += lbuf[row][s] * rg[(size_t)s * kH];
      q2h[row][hh] = (_Float16)acc0;
    }
    __syncthreads();

    // ---- pointer query projection: q = q2 @ pWq^T + pbq ----
    {
      const _Float16* Wq = pWq16 + opaque_zero();
#pragma unroll
      for (int tt = 0; tt < 2; ++tt) {
        const int id = wave * 2 + tt, m = id >> 3, j = id & 7;
        v8f aq = vzero;
#pragma unroll
        for (int kb = 0; kb < 4; ++kb) {
          v16h a = load_frag(&q2h[0][0], kH, m * 16, kb * 32, lane);
          v16h b = load_frag(Wq, kH, j * 16, kb * 32, lane);
          aq = wmma16(a, b, aq);
        }
        const int jc = j * 16 + (lane & 15);
#pragma unroll
        for (int v = 0; v < 8; ++v)
          qbuf[m * 16 + v + ((lane >> 4) << 3)][jc] = aq[v] + pbq[jc];
      }
    }
    __syncthreads();

    // ---- pointer logits, 10*tanh, mask ----
    for (int p = tid; p < kBM * kS; p += 256) {
      const int row = p & (kBM - 1), s = p >> 5;
      const float* rp = refp + ((size_t)(b0 + row) * kS + s) * kH;
      float acc0 = 0.f;
#pragma unroll 8
      for (int hh = 0; hh < kH; ++hh)
        acc0 += sVp[hh] * fast_tanh(qbuf[row][hh] + rp[hh]);
      acc0 = kTanhC * fast_tanh(acc0);
      lbuf[row][s] = maskb[row][s] ? -3.0e38f : acc0;
    }
    __syncthreads();

    if (tid == 0) { localflag = 0; gconf = 0; }
    __syncthreads();

    // ---- softmax -> probs out, sample, per-column conflict scan ----
    if (tid < kBM) {
      float mx = -3.0e38f;
      for (int s = 0; s < kS; ++s) mx = fmaxf(mx, lbuf[tid][s]);
      float sum = 0.f;
      for (int s = 0; s < kS; ++s) {
        float e = __expf(lbuf[tid][s] - mx); lbuf[tid][s] = e; sum += e;
      }
      float inv = fast_rcp(sum);
      float* po = probs_out + ((size_t)t * kB + (b0 + tid)) * kS;
      for (int s = 0; s < kS; ++s) { lbuf[tid][s] *= inv; po[s] = lbuf[tid][s]; }
      // sample
      float u = rng01((uint32_t)t, (uint32_t)(b0 + tid), 0x1234567u);
      myidx = kS - 1; float cacc = 0.f;
      for (int s = 0; s < kS; ++s) {
        cacc += lbuf[tid][s];
        if (u < cacc) { myidx = s; break; }
      }
      // conflict: any hist[t'][b] == idx[b] (column-wise, matching reference)
      int conf = 0;
      for (int t2 = 0; t2 < t; ++t2) conf |= (hist[t2][tid] == (short)myidx);
      if (conf) atomicOr(&localflag, 1);
    }
    __syncthreads();
    if (tid == 0 && localflag) atomicOr(&gflags[t], 1);
    // cluster-wide rendezvous (NOP when not dispatched as a cluster)
    __builtin_amdgcn_s_cluster_barrier();
    if (tid == 0) gconf = atomicOr(&gflags[t], 0);
    __syncthreads();
    if (tid < kBM) {
      if (gconf) {  // resample once for all rows, like the reference
        float u = rng01((uint32_t)t, (uint32_t)(b0 + tid), 0x89ABCDEu);
        myidx = kS - 1; float cacc = 0.f;
        for (int s = 0; s < kS; ++s) {
          cacc += lbuf[tid][s];
          if (u < cacc) { myidx = s; break; }
        }
      }
      hist[t][tid] = (short)myidx;
      curidx[tid] = myidx;
      idx_out[(size_t)t * kB + (b0 + tid)] = (float)myidx;
    }
    __syncthreads();
    // ---- next dec_in = embedded[b, idx] ----
    for (int p = tid; p < kBM * kH; p += 256) {
      const int row = p >> 7, e = p & (kH - 1);
      xbuf[row][e] = emb16[((size_t)(b0 + row) * kS + curidx[row]) * kE + e];
    }
    __syncthreads();
  }
}

// ---------------- host launch ----------------
extern "C" void kernel_launch(void* const* d_in, const int* in_sizes, int n_in,
                              void* d_out, int out_size, void* d_ws, size_t ws_size,
                              hipStream_t stream) {
  const float* batch_input = (const float*)d_in[0];
  const float* emb      = (const float*)d_in[1];
  const float* enc_Wih  = (const float*)d_in[2];
  const float* enc_Whh  = (const float*)d_in[3];
  const float* enc_bih  = (const float*)d_in[4];
  const float* enc_bhh  = (const float*)d_in[5];
  const float* dec_Wih  = (const float*)d_in[6];
  const float* dec_Whh  = (const float*)d_in[7];
  const float* dec_bih  = (const float*)d_in[8];
  const float* dec_bhh  = (const float*)d_in[9];
  const float* g_Wq     = (const float*)d_in[10];
  const float* g_bq     = (const float*)d_in[11];
  const float* g_Wref   = (const float*)d_in[12];
  const float* g_bref   = (const float*)d_in[13];
  const float* g_V      = (const float*)d_in[14];
  const float* p_Wq     = (const float*)d_in[15];
  const float* p_bq     = (const float*)d_in[16];
  const float* p_Wref   = (const float*)d_in[17];
  const float* p_bref   = (const float*)d_in[18];
  const float* p_V      = (const float*)d_in[19];
  const float* dec_start= (const float*)d_in[20];

  char* ws = (char*)d_ws;
  size_t off = 0;
  auto alloc = [&](size_t bytes) -> void* {
    void* p = ws + off;
    off = (off + bytes + 255) & ~(size_t)255;
    return p;
  };
  _Float16* emb16     = (_Float16*)alloc((size_t)kB * kS * kE * 2);
  _Float16* enc16     = (_Float16*)alloc((size_t)kB * kS * kH * 2);
  float*    refg      = (float*)   alloc((size_t)kB * kS * kH * 4);
  float*    refp      = (float*)   alloc((size_t)kB * kS * kH * 4);
  float*    hN        = (float*)   alloc((size_t)kB * kH * 4);
  float*    cN        = (float*)   alloc((size_t)kB * kH * 4);
  _Float16* encWih16  = (_Float16*)alloc((size_t)4 * kH * kE * 2);
  _Float16* encWhh16  = (_Float16*)alloc((size_t)4 * kH * kH * 2);
  _Float16* decWih16  = (_Float16*)alloc((size_t)4 * kH * kE * 2);
  _Float16* decWhh16  = (_Float16*)alloc((size_t)4 * kH * kH * 2);
  _Float16* gWq16     = (_Float16*)alloc((size_t)kH * kH * 2);
  _Float16* gWref16   = (_Float16*)alloc((size_t)kH * kH * 2);
  _Float16* pWq16     = (_Float16*)alloc((size_t)kH * kH * 2);
  _Float16* pWref16   = (_Float16*)alloc((size_t)kH * kH * 2);
  _Float16* start16   = (_Float16*)alloc((size_t)kE * 2);
  int*      gflags    = (int*)     alloc((size_t)kS * 4);
  (void)in_sizes; (void)n_in; (void)out_size; (void)ws_size;

  auto conv = [&](const float* s, _Float16* d, int n) {
    f32_to_f16_kernel<<<(n + 255) / 256, 256, 0, stream>>>(s, d, n);
  };
  conv(enc_Wih, encWih16, 4 * kH * kE);
  conv(enc_Whh, encWhh16, 4 * kH * kH);
  conv(dec_Wih, decWih16, 4 * kH * kE);
  conv(dec_Whh, decWhh16, 4 * kH * kH);
  conv(g_Wq,   gWq16,   kH * kH);
  conv(g_Wref, gWref16, kH * kH);
  conv(p_Wq,   pWq16,   kH * kH);
  conv(p_Wref, pWref16, kH * kH);
  conv(dec_start, start16, kE);

  embed_kernel<<<(kB * kS * kE + 255) / 256, 256, 0, stream>>>(batch_input, emb, emb16);
  zero_flags_kernel<<<1, 64, 0, stream>>>(gflags, kS);

  enc_lstm_kernel<<<kB / kBM, 256, 0, stream>>>(emb16, encWih16, encWhh16,
                                                enc_bih, enc_bhh, enc16, hN, cN);

  refproj_kernel<<<(kB * kS) / kBM, 256, 0, stream>>>(enc16, gWref16, g_bref,
                                                      pWref16, p_bref, refg, refp);

  dec_kernel<<<kB / kBM, 256, 0, stream>>>(
      emb16, decWih16, decWhh16, dec_bih, dec_bhh,
      gWq16, g_bq, g_V, pWq16, p_bq, p_V,
      refg, refp, hN, cN, start16, gflags, (float*)d_out);
}
